// ConditionalGraphNetwork_5428838662517
// MI455X (gfx1250) — compile-verified, matrix operand order checked by reading, not run on hardware
//
#include <hip/hip_runtime.h>

typedef __attribute__((ext_vector_type(16))) __bf16 bvec;   // 8 VGPRs: A or B fragment
typedef __attribute__((ext_vector_type(8)))  float  fvec8;  // 8 VGPRs: C/D fragment

#define Hdim 128
#define NT 8   // 128 / 16 output tiles

__device__ __forceinline__ fvec8 wmma_bf16(bvec a, bvec b, fvec8 c) {
  return __builtin_amdgcn_wmma_f32_16x16x32_bf16(false, a, false, b, (short)0, c, false, false);
}

__device__ __forceinline__ fvec8 fzero8() {
  fvec8 z = {0.f, 0.f, 0.f, 0.f, 0.f, 0.f, 0.f, 0.f};
  return z;
}

// ---------------------------------------------------------------------------
// Weight packing: W [K,128] fp32 row-major -> bf16 B-fragment layout.
// Fragment element (ks, nt, lane, j): W[ks*32 + (lane>=16)*16 + j][nt*16 + (lane&15)]
// stored at dst[((ks*8+nt)*32+lane)*16 + j]. K padded with zeros to Ksteps*32.
// ---------------------------------------------------------------------------
__global__ void pack_kernel(const float* __restrict__ W, __bf16* __restrict__ dst,
                            int Kin, int Ksteps) {
  int t = blockIdx.x * blockDim.x + threadIdx.x;
  int total = Ksteps * 8 * 32;
  if (t >= total) return;
  int lane = t & 31;
  int nt   = (t >> 5) & 7;
  int ks   = t >> 8;
  int colB = nt * 16 + (lane & 15);
  int kB   = ks * 32 + (lane >> 4) * 16;
  __bf16* o = dst + (size_t)t * 16;
#pragma unroll
  for (int j = 0; j < 16; ++j) {
    int k = kB + j;
    o[j] = (k < Kin) ? (__bf16)W[(size_t)k * Hdim + colB] : (__bf16)0.f;
  }
}

__global__ void zero_kernel(float* a, int n) {
  int i = blockIdx.x * blockDim.x + threadIdx.x;
  if (i < n) a[i] = 0.f;
}

// edge_batch = batch[row[e]], edge counts per source node (row is static)
__global__ void prep_kernel(const int* __restrict__ row, const int* __restrict__ batch,
                            int* __restrict__ ebat, float* cnt, int E_) {
  int e = blockIdx.x * blockDim.x + threadIdx.x;
  if (e < E_) {
    int r = row[e];
    ebat[e] = batch[r];
    atomicAdd(&cnt[r], 1.0f);
  }
}

// aggb = bf16(agg / max(cnt,1));  agg reset for next layer
__global__ void mean_kernel(float* __restrict__ agg, const float* __restrict__ cnt,
                            __bf16* __restrict__ aggb, int N_) {
  int i = blockIdx.x * blockDim.x + threadIdx.x;
  if (i < N_ * Hdim) {
    float c = cnt[i >> 7];
    c = c > 1.f ? c : 1.f;
    aggb[i] = (__bf16)(agg[i] / c);
    agg[i] = 0.f;
  }
}

// ---------------------------------------------------------------------------
// Encoder MLP: in [M,Kin] fp32 (Kin<=16 padded to one K-step) -> relu(in@W1+b1)@W2+b2
// outputs bf16 (always) and fp32 (optional).
// One wave = 16 rows x 128 cols.
// ---------------------------------------------------------------------------
__global__ __launch_bounds__(128) void enc_kernel(
    const float* __restrict__ in, int Kin, int M,
    const __bf16* __restrict__ w1p, const float* __restrict__ b1,
    const __bf16* __restrict__ w2p, const float* __restrict__ b2,
    __bf16* __restrict__ outb, float* __restrict__ outf) {
  __shared__ __align__(32) __bf16 hbuf[4][16][Hdim];
  const int lane = threadIdx.x & 31;
  const int wv = threadIdx.x >> 5;
  const int row0 = (blockIdx.x * 4 + wv) * 16;
  const int lm = lane & 15;
  const int kh = (lane >> 4) * 16;
  int m = row0 + lm;
  int mc = (m < M) ? m : (M - 1);

  bvec a;
#pragma unroll
  for (int t = 0; t < 16; ++t) {
    int k = kh + t;
    a[t] = (k < Kin) ? (__bf16)in[(size_t)mc * Kin + k] : (__bf16)0.f;
  }
  fvec8 acc[NT];
#pragma unroll
  for (int nt = 0; nt < NT; ++nt) acc[nt] = fzero8();
#pragma unroll
  for (int nt = 0; nt < NT; ++nt) {
    bvec b = *(const bvec*)(w1p + ((size_t)nt * 32 + lane) * 16);
    acc[nt] = wmma_bf16(a, b, acc[nt]);
  }
  const int rb = kh ? 8 : 0;
#pragma unroll
  for (int nt = 0; nt < NT; ++nt) {
    float bias = b1[nt * 16 + lm];
#pragma unroll
    for (int i = 0; i < 8; ++i) {
      float v = acc[nt][i] + bias;
      hbuf[wv][rb + i][nt * 16 + lm] = (__bf16)(v > 0.f ? v : 0.f);
    }
  }
  fvec8 acc2[NT];
#pragma unroll
  for (int nt = 0; nt < NT; ++nt) acc2[nt] = fzero8();
#pragma unroll
  for (int ks = 0; ks < 4; ++ks) {
    bvec ah = *(const bvec*)&hbuf[wv][lm][ks * 32 + kh];
#pragma unroll
    for (int nt = 0; nt < NT; ++nt) {
      bvec b = *(const bvec*)(w2p + ((size_t)(ks * NT + nt) * 32 + lane) * 16);
      acc2[nt] = wmma_bf16(ah, b, acc2[nt]);
    }
  }
#pragma unroll
  for (int nt = 0; nt < NT; ++nt) {
    float bias = b2[nt * 16 + lm];
#pragma unroll
    for (int i = 0; i < 8; ++i)
      hbuf[wv][rb + i][nt * 16 + lm] = (__bf16)(acc2[nt][i] + bias);
  }
  // write out: lane owns half a row (64 cols)
  const int rr = lane >> 1;
  const int half = (lane & 1) * 64;
  const int rg = row0 + rr;
  if (rg < M) {
    const uint4* src = (const uint4*)&hbuf[wv][rr][half];
    uint4* dst = (uint4*)(outb + (size_t)rg * Hdim + half);
#pragma unroll
    for (int j = 0; j < 8; ++j) dst[j] = src[j];
    if (outf) {
      const __bf16* hv = &hbuf[wv][rr][half];
      float* fo = outf + (size_t)rg * Hdim + half;
#pragma unroll
      for (int j = 0; j < 64; ++j) fo[j] = (float)hv[j];
    }
  }
}

// ---------------------------------------------------------------------------
// Edge layer: e_new = MLP_512( [xb[row] | xb[col] | eb | ub[ebat]] ), in-place on eb,
// plus atomic scatter-add of e_new into agg[row].
// ---------------------------------------------------------------------------
__global__ __launch_bounds__(128) void edge_layer_kernel(
    const __bf16* __restrict__ xb, __bf16* eb, const __bf16* __restrict__ ub,
    const int* __restrict__ row, const int* __restrict__ col, const int* __restrict__ ebat,
    const __bf16* __restrict__ w1p, const float* __restrict__ b1,
    const __bf16* __restrict__ w2p, const float* __restrict__ b2,
    float* agg, int E_) {
  __shared__ __align__(32) __bf16 hbuf[4][16][Hdim];
  const int lane = threadIdx.x & 31;
  const int wv = threadIdx.x >> 5;
  const int edge0 = (blockIdx.x * 4 + wv) * 16;
  const int lm = lane & 15;
  const int kh = (lane >> 4) * 16;
  int m = edge0 + lm;
  if (m >= E_) m = E_ - 1;
  const __bf16* p0 = xb + (size_t)row[m] * Hdim;
  const __bf16* p1 = xb + (size_t)col[m] * Hdim;
  const __bf16* p2 = eb + (size_t)m * Hdim;
  const __bf16* p3 = ub + (size_t)ebat[m] * Hdim;

  fvec8 acc[NT];
#pragma unroll
  for (int nt = 0; nt < NT; ++nt) acc[nt] = fzero8();
#pragma unroll
  for (int ks = 0; ks < 16; ++ks) {
    const __bf16* sp = (ks < 4) ? p0 : (ks < 8) ? p1 : (ks < 12) ? p2 : p3;
    bvec a = *(const bvec*)(sp + ((ks & 3) * 32 + kh));
#pragma unroll
    for (int nt = 0; nt < NT; ++nt) {
      bvec b = *(const bvec*)(w1p + ((size_t)(ks * NT + nt) * 32 + lane) * 16);
      acc[nt] = wmma_bf16(a, b, acc[nt]);
    }
  }
  const int rb = kh ? 8 : 0;
#pragma unroll
  for (int nt = 0; nt < NT; ++nt) {
    float bias = b1[nt * 16 + lm];
#pragma unroll
    for (int i = 0; i < 8; ++i) {
      float v = acc[nt][i] + bias;
      hbuf[wv][rb + i][nt * 16 + lm] = (__bf16)(v > 0.f ? v : 0.f);
    }
  }
  fvec8 acc2[NT];
#pragma unroll
  for (int nt = 0; nt < NT; ++nt) acc2[nt] = fzero8();
#pragma unroll
  for (int ks = 0; ks < 4; ++ks) {
    bvec ah = *(const bvec*)&hbuf[wv][lm][ks * 32 + kh];
#pragma unroll
    for (int nt = 0; nt < NT; ++nt) {
      bvec b = *(const bvec*)(w2p + ((size_t)(ks * NT + nt) * 32 + lane) * 16);
      acc2[nt] = wmma_bf16(ah, b, acc2[nt]);
    }
  }
#pragma unroll
  for (int nt = 0; nt < NT; ++nt) {
    float bias = b2[nt * 16 + lm];
#pragma unroll
    for (int i = 0; i < 8; ++i)
      hbuf[wv][rb + i][nt * 16 + lm] = (__bf16)(acc2[nt][i] + bias);
  }
  const int er = lane >> 1;
  const int half = (lane & 1) * 64;
  const int eg = edge0 + er;
  if (eg < E_) {
    const uint4* src = (const uint4*)&hbuf[wv][er][half];
    uint4* dst = (uint4*)(eb + (size_t)eg * Hdim + half);
#pragma unroll
    for (int j = 0; j < 8; ++j) dst[j] = src[j];
    float* aggp = agg + (size_t)row[eg] * Hdim + half;
    const __bf16* hv = &hbuf[wv][er][half];
#pragma unroll
    for (int j = 0; j < 64; ++j) atomicAdd(aggp + j, (float)hv[j]);
  }
}

// ---------------------------------------------------------------------------
// Node layer: x = x + MLP_384( [xb | aggb | ub[batch]] ); updates x fp32 + xb bf16
// ---------------------------------------------------------------------------
__global__ __launch_bounds__(128) void node_layer_kernel(
    __bf16* xb, const __bf16* __restrict__ aggb, const __bf16* __restrict__ ub,
    const int* __restrict__ batch,
    const __bf16* __restrict__ w1p, const float* __restrict__ b1,
    const __bf16* __restrict__ w2p, const float* __restrict__ b2,
    float* x, int N_) {
  __shared__ __align__(32) __bf16 hbuf[4][16][Hdim];
  const int lane = threadIdx.x & 31;
  const int wv = threadIdx.x >> 5;
  const int node0 = (blockIdx.x * 4 + wv) * 16;
  const int lm = lane & 15;
  const int kh = (lane >> 4) * 16;
  int n = node0 + lm;
  if (n >= N_) n = N_ - 1;
  const __bf16* p0 = xb + (size_t)n * Hdim;
  const __bf16* p1 = aggb + (size_t)n * Hdim;
  const __bf16* p2 = ub + (size_t)batch[n] * Hdim;

  fvec8 acc[NT];
#pragma unroll
  for (int nt = 0; nt < NT; ++nt) acc[nt] = fzero8();
#pragma unroll
  for (int ks = 0; ks < 12; ++ks) {
    const __bf16* sp = (ks < 4) ? p0 : (ks < 8) ? p1 : p2;
    bvec a = *(const bvec*)(sp + ((ks & 3) * 32 + kh));
#pragma unroll
    for (int nt = 0; nt < NT; ++nt) {
      bvec b = *(const bvec*)(w1p + ((size_t)(ks * NT + nt) * 32 + lane) * 16);
      acc[nt] = wmma_bf16(a, b, acc[nt]);
    }
  }
  const int rb = kh ? 8 : 0;
#pragma unroll
  for (int nt = 0; nt < NT; ++nt) {
    float bias = b1[nt * 16 + lm];
#pragma unroll
    for (int i = 0; i < 8; ++i) {
      float v = acc[nt][i] + bias;
      hbuf[wv][rb + i][nt * 16 + lm] = (__bf16)(v > 0.f ? v : 0.f);
    }
  }
  fvec8 acc2[NT];
#pragma unroll
  for (int nt = 0; nt < NT; ++nt) acc2[nt] = fzero8();
#pragma unroll
  for (int ks = 0; ks < 4; ++ks) {
    bvec ah = *(const bvec*)&hbuf[wv][lm][ks * 32 + kh];
#pragma unroll
    for (int nt = 0; nt < NT; ++nt) {
      bvec b = *(const bvec*)(w2p + ((size_t)(ks * NT + nt) * 32 + lane) * 16);
      acc2[nt] = wmma_bf16(ah, b, acc2[nt]);
    }
  }
#pragma unroll
  for (int nt = 0; nt < NT; ++nt) {
    float bias = b2[nt * 16 + lm];
#pragma unroll
    for (int i = 0; i < 8; ++i)
      hbuf[wv][rb + i][nt * 16 + lm] = (__bf16)(acc2[nt][i] + bias);
  }
  const int nr = lane >> 1;
  const int half = (lane & 1) * 64;
  const int ng = node0 + nr;
  if (ng < N_) {
    const __bf16* hv = &hbuf[wv][nr][half];
    float* xp = x + (size_t)ng * Hdim + half;
    __bf16* xbp = xb + (size_t)ng * Hdim + half;
#pragma unroll
    for (int j = 0; j < 64; ++j) {
      float v = (float)hv[j] + xp[j];   // residual in fp32
      xp[j] = v;
      xbp[j] = (__bf16)v;
    }
  }
}

// Decoder 128 -> 128 -> 3 (tiny; plain VALU)
__global__ __launch_bounds__(128) void dec_kernel(
    const float* __restrict__ x, const float* __restrict__ w1, const float* __restrict__ b1,
    const float* __restrict__ w2, const float* __restrict__ b2, float* __restrict__ out, int N_) {
  __shared__ float xr[Hdim];
  __shared__ float hr[Hdim];
  int n = blockIdx.x;
  int j = threadIdx.x;
  if (n >= N_) return;
  xr[j] = x[(size_t)n * Hdim + j];
  __syncthreads();
  float h = b1[j];
  for (int k = 0; k < Hdim; ++k) h += xr[k] * w1[k * Hdim + j];
  hr[j] = h > 0.f ? h : 0.f;
  __syncthreads();
  if (j < 3) {
    float o = b2[j];
    for (int k = 0; k < Hdim; ++k) o += hr[k] * w2[k * 3 + j];
    out[(size_t)n * 3 + j] = o;
  }
}

// ---------------------------------------------------------------------------
extern "C" void kernel_launch(void* const* d_in, const int* in_sizes, int n_in,
                              void* d_out, int out_size, void* d_ws, size_t ws_size,
                              hipStream_t stream) {
  (void)in_sizes; (void)n_in; (void)out_size; (void)ws_size;
  const int N_ = 50000, E_ = 400000, B_ = 8;
  const float* x_in  = (const float*)d_in[0];
  const float* eattr = (const float*)d_in[1];
  const float* cond  = (const float*)d_in[2];
  // params in insertion order: node_enc(3..6) edge_enc(7..10) cond_enc(11..14)
  // layers l: edge w1,b1,w2,b2 = 15+8l..  node w1,b1,w2,b2 = 19+8l..  dec(47..50)
  const int* eidx  = (const int*)d_in[51];
  const int* batch = (const int*)d_in[52];
  const int* row = eidx;
  const int* col = eidx + E_;
  auto P = [&](int i) { return (const float*)d_in[i]; };

  char* base = (char*)d_ws;
  size_t off = 0;
  auto carve = [&](size_t bytes) -> void* {
    void* p = base + off;
    off = (off + bytes + 255) & ~(size_t)255;
    return p;
  };
  float*  x    = (float*)carve((size_t)N_ * Hdim * 4);
  __bf16* xb   = (__bf16*)carve((size_t)N_ * Hdim * 2);
  __bf16* eb   = (__bf16*)carve((size_t)E_ * Hdim * 2);
  __bf16* ub   = (__bf16*)carve((size_t)B_ * Hdim * 2);
  float*  agg  = (float*)carve((size_t)N_ * Hdim * 4);
  __bf16* aggb = (__bf16*)carve((size_t)N_ * Hdim * 2);
  float*  cnt  = (float*)carve((size_t)N_ * 4);
  int*    ebat = (int*)carve((size_t)E_ * 4);
  auto carveb = [&](size_t elems) { return (__bf16*)carve(elems * 2); };
  __bf16* encNw1 = carveb(32 * 128);  __bf16* encNw2 = carveb(128 * 128);
  __bf16* encEw1 = carveb(32 * 128);  __bf16* encEw2 = carveb(128 * 128);
  __bf16* encCw1 = carveb(32 * 128);  __bf16* encCw2 = carveb(128 * 128);
  __bf16 *Ew1[4], *Ew2[4], *Nw1[4], *Nw2[4];
  for (int l = 0; l < 4; ++l) {
    Ew1[l] = carveb(512 * 128);  Ew2[l] = carveb(128 * 128);
    Nw1[l] = carveb(384 * 128);  Nw2[l] = carveb(128 * 128);
  }

  auto pack = [&](const float* W, __bf16* dst, int Kin, int Ks) {
    int total = Ks * 8 * 32;
    pack_kernel<<<(total + 127) / 128, 128, 0, stream>>>(W, dst, Kin, Ks);
  };
  pack(P(3), encNw1, 16, 1);   pack(P(5), encNw2, 128, 4);
  pack(P(7), encEw1, 4, 1);    pack(P(9), encEw2, 128, 4);
  pack(P(11), encCw1, 8, 1);   pack(P(13), encCw2, 128, 4);
  for (int l = 0; l < 4; ++l) {
    int b0 = 15 + l * 8;
    pack(P(b0 + 0), Ew1[l], 512, 16);  pack(P(b0 + 2), Ew2[l], 128, 4);
    pack(P(b0 + 4), Nw1[l], 384, 12);  pack(P(b0 + 6), Nw2[l], 128, 4);
  }

  zero_kernel<<<(N_ * Hdim + 255) / 256, 256, 0, stream>>>(agg, N_ * Hdim);
  zero_kernel<<<(N_ + 255) / 256, 256, 0, stream>>>(cnt, N_);
  prep_kernel<<<(E_ + 255) / 256, 256, 0, stream>>>(row, batch, ebat, cnt, E_);

  enc_kernel<<<(N_ + 63) / 64, 128, 0, stream>>>(x_in, 16, N_, encNw1, P(4), encNw2, P(6), xb, x);
  enc_kernel<<<(E_ + 63) / 64, 128, 0, stream>>>(eattr, 4, E_, encEw1, P(8), encEw2, P(10), eb, (float*)nullptr);
  enc_kernel<<<1, 128, 0, stream>>>(cond, 8, B_, encCw1, P(12), encCw2, P(14), ub, (float*)nullptr);

  for (int l = 0; l < 4; ++l) {
    int b0 = 15 + l * 8;
    edge_layer_kernel<<<(E_ + 63) / 64, 128, 0, stream>>>(
        xb, eb, ub, row, col, ebat, Ew1[l], P(b0 + 1), Ew2[l], P(b0 + 3), agg, E_);
    mean_kernel<<<(N_ * Hdim + 255) / 256, 256, 0, stream>>>(agg, cnt, aggb, N_);
    node_layer_kernel<<<(N_ + 63) / 64, 128, 0, stream>>>(
        xb, aggb, ub, batch, Nw1[l], P(b0 + 5), Nw2[l], P(b0 + 7), x, N_);
  }
  dec_kernel<<<N_, 128, 0, stream>>>(x, P(47), P(48), P(49), P(50), (float*)d_out, N_);
}
